// CNN_66889820668517
// MI455X (gfx1250) — matrix operand order repaired and checked by reading, not compiled, not run on hardware
//
#include <hip/hip_runtime.h>

// ---------------------------------------------------------------------------
// CDNA5 (gfx1250) SPP-CNN forward pass.
// GEMM work (ten 2x2 convs + three FC layers) on v_wmma_f32_16x16x32_f16
// (f16 A/B, f32 accumulate). Activations NHWC f16.
// Conv GEMM: weight tiles double-buffered in LDS via GLOBAL_LOAD_ASYNC_TO_LDS
// (s_wait_asynccnt + barrier), fully unrolled K pipeline with compile-time
// B offsets (kernel templated on log2(Cin)).
// ---------------------------------------------------------------------------

typedef __attribute__((ext_vector_type(16))) _Float16 v16h;
typedef __attribute__((ext_vector_type(8)))  float    v8f;

union HFrag { v16h v; uint4 q[2]; };
union H8    { _Float16 h[8]; uint4 q; };

__device__ __forceinline__ uint4 ldq16(const _Float16* p) {
    return *reinterpret_cast<const uint4*>(p);
}

__device__ __forceinline__ v8f wmma16(const HFrag& a, const HFrag& b, v8f c) {
    // D(16x16,f32) = A(16x32,f16) * B(32x16,f16) + C
    return __builtin_amdgcn_wmma_f32_16x16x32_f16(false, a.v, false, b.v,
                                                  (short)0, c, false, false);
}

// Async global->LDS copy (gfx1250, tracked by ASYNCcnt).
__device__ __forceinline__ void async_copy_b128(unsigned lds_byte_addr,
                                                const _Float16* g) {
    asm volatile("global_load_async_to_lds_b128 %0, %1, off"
                 :: "v"(lds_byte_addr), "v"(g)
                 : "memory");
}
__device__ __forceinline__ void wait_asynccnt0() {
    asm volatile("s_wait_asynccnt 0x0" ::: "memory");
}

// ---------------------------------------------------------------------------
// conv1: 7x7 stride2 pad3, Cin=3, Cout=64, fp32 math, NCHW in -> NHWC f16 out.
// ---------------------------------------------------------------------------
__global__ void __launch_bounds__(256)
conv1_k(const float* __restrict__ x, const float* __restrict__ w,
        const float* __restrict__ bias, _Float16* __restrict__ out)
{
    __shared__ float wl[64 * 147];
    for (int i = threadIdx.x; i < 64 * 147; i += 256) wl[i] = w[i];
    __syncthreads();

    int gid = blockIdx.x * 256 + threadIdx.x;      // 64*112*112 pixels
    int b  = gid / (112 * 112);
    int rm = gid - b * 112 * 112;
    int oy = rm / 112;
    int ox = rm - oy * 112;

    float acc[64];
#pragma unroll
    for (int co = 0; co < 64; ++co) acc[co] = bias[co];

    for (int ci = 0; ci < 3; ++ci) {
        for (int ky = 0; ky < 7; ++ky) {
            int iy = oy * 2 - 3 + ky;
            if (iy < 0 || iy >= 224) continue;
            for (int kx = 0; kx < 7; ++kx) {
                int ix = ox * 2 - 3 + kx;
                if (ix < 0 || ix >= 224) continue;
                float v = x[(((long)b * 3 + ci) * 224 + iy) * 224 + ix];
                int idx = (ci * 7 + ky) * 7 + kx;
#pragma unroll
                for (int co = 0; co < 64; ++co)
                    acc[co] = fmaf(v, wl[co * 147 + idx], acc[co]);
            }
        }
    }

    long obase = (long)gid * 64;
#pragma unroll
    for (int cg = 0; cg < 8; ++cg) {
        H8 o;
#pragma unroll
        for (int v = 0; v < 8; ++v) {
            float f = acc[cg * 8 + v];
            o.h[v] = (_Float16)(f > 0.f ? f : 0.f);
        }
        *reinterpret_cast<uint4*>(out + obase + cg * 8) = o.q;
    }
}

// ---------------------------------------------------------------------------
// NHWC max pool (VALID), f16.
// ---------------------------------------------------------------------------
__global__ void __launch_bounds__(256)
maxpool_k(const _Float16* __restrict__ in, _Float16* __restrict__ out,
          int Hi, int Wi, int Ho, int Wo, int C, int win, int stride, int total)
{
    int gid = blockIdx.x * blockDim.x + threadIdx.x;
    if (gid >= total) return;
    int c = gid % C;
    int t = gid / C;
    int x = t % Wo; t /= Wo;
    int y = t % Ho;
    int b = t / Ho;
    float m = -3.4e38f;
    for (int ky = 0; ky < win; ++ky)
        for (int kx = 0; kx < win; ++kx) {
            int iy = y * stride + ky, ix = x * stride + kx;
            float v = (float)in[(((long)b * Hi + iy) * Wi + ix) * C + c];
            m = v > m ? v : m;
        }
    out[gid] = (_Float16)m;
}

// ---------------------------------------------------------------------------
// Reorder 2x2 conv weights OIHW fp32 -> [Cout][dydx][Cin] f16.
// ---------------------------------------------------------------------------
__global__ void __launch_bounds__(256)
reorder_w2x2(const float* __restrict__ src, _Float16* __restrict__ dst,
             int Cout, int Cin, int total)
{
    int gid = blockIdx.x * blockDim.x + threadIdx.x;
    if (gid >= total) return;
    int co   = gid / (Cin * 4);
    int rem  = gid - co * Cin * 4;
    int dydx = rem / Cin;
    int ci   = rem - dydx * Cin;
    dst[gid] = (_Float16)src[((long)co * Cin + ci) * 4 + dydx];
}

__global__ void __launch_bounds__(256)
cvt_f32_f16(const float* __restrict__ src, _Float16* __restrict__ dst, long n)
{
    long i = (long)blockIdx.x * blockDim.x + threadIdx.x;
    if (i < n) dst[i] = (_Float16)src[i];
}

// ---------------------------------------------------------------------------
// 2x2 conv as WMMA GEMM, templated on log2(Cin) so every B offset is a
// compile-time constant (folds into the global_load offset field).
// D[co, pix] = sum_k W[co,k] * im2col[k,pix] + bias;  k = (dy*2+dx)*Cin + ci.
// Block = 8 waves sharing one 32-row M tile, covering 8 N tiles of 32 pixels.
// Weight tile double-buffered in LDS via async global->LDS DMA.
// ---------------------------------------------------------------------------
#define KCH 128

template<int CINSHIFT>
__global__ void __launch_bounds__(256)
conv2x2_gemm_t(const _Float16* __restrict__ act, const _Float16* __restrict__ wA,
               const float* __restrict__ bias, _Float16* __restrict__ out,
               int Hi, int Wi, int Cout, int Ho, int Wo, int nTiles)
{
    constexpr int Cin   = 1 << CINSHIFT;
    constexpr int cmask = Cin - 1;
    constexpr int K     = Cin << 2;
    constexpr int nch   = K / KCH;

    __shared__ _Float16 shA[2][32 * KCH];   // 2 x 8KB double buffer

    const int tid   = threadIdx.x;
    const int lane  = tid & 31;
    const int wv    = tid >> 5;
    const int mTiles = (Cout + 31) >> 5;
    const int blkM = blockIdx.x % mTiles;
    const int blkN = blockIdx.x / mTiles;
    int ni = blkN * 8 + wv;
    if (ni >= nTiles) ni = nTiles - 1;   // tail waves duplicate last tile (uniform barriers)
    const int r  = lane & 15;
    const int kh = lane >> 4;
    const int mBase = blkM << 5;
    const int nBase = ni << 5;
    const int HoWo  = Ho * Wo;

    // per-wave column base pointers (two 16-column groups)
    const _Float16* bp[2];
    int pix[2];
#pragma unroll
    for (int j = 0; j < 2; ++j) {
        int n = nBase + j * 16 + r;
        int b = n / HoWo;
        int rem = n - b * HoWo;
        int y = rem / Wo;
        int x = rem - y * Wo;
        pix[j] = n;
        bp[j]  = act + ((((long)b * Hi + y) * Wi + x) << CINSHIFT);
    }
    const long rowStride = (long)Wi << CINSHIFT;   // dy=1 element offset

    // Async A staging: 32 rows x KCH halves = 512 b128 chunks, 2 per thread.
    // Chunk q -> tile row q>>4, k offset (q&15)*8; LDS offset = q*16 bytes.
    const int q0 = tid * 2, q1 = q0 + 1;
    int arow0 = mBase + (q0 >> 4); if (arow0 >= Cout) arow0 = Cout - 1;
    int arow1 = mBase + (q1 >> 4); if (arow1 >= Cout) arow1 = Cout - 1;
    const _Float16* gA0 = wA + (long)arow0 * K + ((q0 & 15) << 3);
    const _Float16* gA1 = wA + (long)arow1 * K + ((q1 & 15) << 3);
    const unsigned lb0 = (unsigned)(size_t)&shA[0][0];
    const unsigned lb1 = (unsigned)(size_t)&shA[1][0];
    const unsigned o0 = (unsigned)q0 * 16u, o1 = (unsigned)q1 * 16u;

    v8f acc[2][2] = {};

    // prime buffer 0
    async_copy_b128(lb0 + o0, gA0);
    async_copy_b128(lb0 + o1, gA1);

#pragma unroll
    for (int c = 0; c < nch; ++c) {
        wait_asynccnt0();
        __syncthreads();                       // chunk c resident in buf c&1
        if (c + 1 < nch) {                     // overlap DMA of next chunk
            unsigned lb = ((c + 1) & 1) ? lb1 : lb0;
            const int kk = (c + 1) * KCH;
            async_copy_b128(lb + o0, gA0 + kk);
            async_copy_b128(lb + o1, gA1 + kk);
        }
        const _Float16* As = &shA[c & 1][0];

#pragma unroll
        for (int kl = 0; kl < KCH; kl += 32) {
            const int kc0 = kl + (kh << 3);    // A chunk 0 (8 halves)
            const int kc1 = kc0 + 16;          // A chunk 1 (8 halves)
            const int kb  = c * KCH + kl + (kh << 3);   // B: 8 halves at kb, 8 at kb+16

            HFrag a0, a1, b0, b1;
            a0.q[0] = *reinterpret_cast<const uint4*>(As + r * KCH + kc0);
            a0.q[1] = *reinterpret_cast<const uint4*>(As + r * KCH + kc1);
            a1.q[0] = *reinterpret_cast<const uint4*>(As + (16 + r) * KCH + kc0);
            a1.q[1] = *reinterpret_cast<const uint4*>(As + (16 + r) * KCH + kc1);

            // compile-time except for the kh-dependent part folded in kb
            const int  dydx = kb >> CINSHIFT;  // 0..3
            const long boff = (long)(dydx >> 1) * rowStride
                            + (long)((dydx & 1) << CINSHIFT) + (kb & cmask);
            b0.q[0] = ldq16(bp[0] + boff);
            b0.q[1] = ldq16(bp[0] + boff + 16);
            b1.q[0] = ldq16(bp[1] + boff);
            b1.q[1] = ldq16(bp[1] + boff + 16);

            acc[0][0] = wmma16(a0, b0, acc[0][0]);
            acc[0][1] = wmma16(a0, b1, acc[0][1]);
            acc[1][0] = wmma16(a1, b0, acc[1][0]);
            acc[1][1] = wmma16(a1, b1, acc[1][1]);
        }
    }

#pragma unroll
    for (int jm = 0; jm < 2; ++jm) {
        const int mo = mBase + jm * 16 + (kh << 3);
        float bs[8];
#pragma unroll
        for (int v = 0; v < 8; ++v) bs[v] = bias[mo + v];
#pragma unroll
        for (int jn = 0; jn < 2; ++jn) {
            H8 o;
#pragma unroll
            for (int v = 0; v < 8; ++v) {
                float f = acc[jm][jn][v] + bs[v];
                o.h[v] = (_Float16)(f > 0.f ? f : 0.f);
            }
            *reinterpret_cast<uint4*>(out + (long)pix[jn] * Cout + mo) = o.q;
        }
    }
}

// ---------------------------------------------------------------------------
// FC layer as WMMA GEMM.  out[n, m] = sum_k act[n,k] * W[m,k] + bias[m].
// act: [N,K] f16 row-major; W: [M,K] f16 row-major. N multiple of 32.
// outF32=1 -> fp32 output (final logits, store-guarded for M=1000).
// ---------------------------------------------------------------------------
__global__ void __launch_bounds__(256)
fc_gemm(const _Float16* __restrict__ act, const _Float16* __restrict__ wA,
        const float* __restrict__ bias, void* __restrict__ outPtr,
        int M, int K, int N, int doRelu, int outF32)
{
    const int lane = threadIdx.x & 31;
    const int wid  = blockIdx.x * (blockDim.x >> 5) + (threadIdx.x >> 5);
    const int mTiles = (M + 31) >> 5;
    const int nTiles = N >> 5;
    const int mi = wid % mTiles;
    const int ni = wid / mTiles;
    if (ni >= nTiles) return;
    const int r = lane & 15, kh = lane >> 4;
    const int mBase = mi << 5, nBase = ni << 5;

    const _Float16* b0p = act + (long)(nBase + r) * K;
    const _Float16* b1p = act + (long)(nBase + 16 + r) * K;
    int rowA0 = mBase + r;      if (rowA0 >= M) rowA0 = M - 1;
    int rowA1 = mBase + 16 + r; if (rowA1 >= M) rowA1 = M - 1;
    const _Float16* a0p = wA + (long)rowA0 * K;
    const _Float16* a1p = wA + (long)rowA1 * K;

    v8f acc[2][2] = {};
    for (int kk = 0; kk < K; kk += 32) {
        const int kc0 = kk + (kh << 3);
        const int kc1 = kc0 + 16;
        const int kb  = kk + (kh << 3);
        HFrag a0, a1, b0, b1;
        a0.q[0] = ldq16(a0p + kc0);  a0.q[1] = ldq16(a0p + kc1);
        a1.q[0] = ldq16(a1p + kc0);  a1.q[1] = ldq16(a1p + kc1);
        b0.q[0] = ldq16(b0p + kb);   b0.q[1] = ldq16(b0p + kb + 16);
        b1.q[0] = ldq16(b1p + kb);   b1.q[1] = ldq16(b1p + kb + 16);
        acc[0][0] = wmma16(a0, b0, acc[0][0]);
        acc[0][1] = wmma16(a0, b1, acc[0][1]);
        acc[1][0] = wmma16(a1, b0, acc[1][0]);
        acc[1][1] = wmma16(a1, b1, acc[1][1]);
    }

#pragma unroll
    for (int jm = 0; jm < 2; ++jm) {
        const int mo = mBase + jm * 16 + (kh << 3);
#pragma unroll
        for (int jn = 0; jn < 2; ++jn) {
            const int n = nBase + jn * 16 + r;
            if (outF32) {
                float* out = (float*)outPtr;
#pragma unroll
                for (int v = 0; v < 8; ++v) {
                    int m = mo + v;
                    if (m < M) {
                        float f = acc[jm][jn][v] + bias[m];
                        if (doRelu) f = f > 0.f ? f : 0.f;
                        out[(long)n * M + m] = f;
                    }
                }
            } else {
                _Float16* out = (_Float16*)outPtr;
                H8 o;
#pragma unroll
                for (int v = 0; v < 8; ++v) {
                    float f = acc[jm][jn][v] + bias[mo + v];
                    if (doRelu) f = f > 0.f ? f : 0.f;
                    o.h[v] = (_Float16)f;
                }
                *reinterpret_cast<uint4*>(out + (long)n * M + mo) = o.q;
            }
        }
    }
}

// ---------------------------------------------------------------------------
// Spatial pyramid pooling: NHWC [64,10,10,256] f16 -> [64, c*50+bin] f16,
// bin order = levels (6,3,2,1), PyTorch adaptive bin boundaries.
// ---------------------------------------------------------------------------
__global__ void __launch_bounds__(256)
spp_k(const _Float16* __restrict__ in, _Float16* __restrict__ out)
{
    int gid = blockIdx.x * blockDim.x + threadIdx.x;   // 64*50*256
    if (gid >= 64 * 50 * 256) return;
    int c   = gid % 256;
    int t   = gid / 256;
    int bin = t % 50;
    int b   = t / 50;

    int L, idx;
    if (bin < 36)      { L = 6; idx = bin; }
    else if (bin < 45) { L = 3; idx = bin - 36; }
    else if (bin < 49) { L = 2; idx = bin - 45; }
    else               { L = 1; idx = 0; }
    int i = idx / L, j = idx % L;
    int r0 = (i * 10) / L, r1 = ((i + 1) * 10 + L - 1) / L;
    int c0 = (j * 10) / L, c1 = ((j + 1) * 10 + L - 1) / L;

    float s = 0.f;
    for (int y = r0; y < r1; ++y)
        for (int xx = c0; xx < c1; ++xx)
            s += (float)in[(((long)b * 10 + y) * 10 + xx) * 256 + c];
    s /= (float)((r1 - r0) * (c1 - c0));
    out[(long)b * 12800 + c * 50 + bin] = (_Float16)s;
}

// ---------------------------------------------------------------------------

static inline int cdiv_l(long a, long b) { return (int)((a + b - 1) / b); }

extern "C" void kernel_launch(void* const* d_in, const int* in_sizes, int n_in,
                              void* d_out, int out_size, void* d_ws, size_t ws_size,
                              hipStream_t stream)
{
    (void)in_sizes; (void)n_in; (void)out_size; (void)ws_size;
    const int B = 64;

    const float* x  = (const float*)d_in[0];
    const float* w1 = (const float*)d_in[1];
    const float* b1 = (const float*)d_in[2];
    const float* w2[4]; const float* b2[4];
    for (int i = 0; i < 4; ++i) { w2[i] = (const float*)d_in[3 + 2*i]; b2[i] = (const float*)d_in[4 + 2*i]; }
    const float* w3[6]; const float* b3[6];
    for (int i = 0; i < 6; ++i) { w3[i] = (const float*)d_in[11 + 2*i]; b3[i] = (const float*)d_in[12 + 2*i]; }
    const float* fc1w = (const float*)d_in[23]; const float* fc1b = (const float*)d_in[24];
    const float* fc2w = (const float*)d_in[25]; const float* fc2b = (const float*)d_in[26];
    const float* fc3w = (const float*)d_in[27]; const float* fc3b = (const float*)d_in[28];

    // bump allocator over d_ws (f16 elements, 256B aligned)
    char* base = (char*)d_ws;
    size_t off = 0;
    auto alloc16 = [&](long n) -> _Float16* {
        _Float16* p = (_Float16*)(base + off);
        off += ((size_t)n * 2 + 255) & ~(size_t)255;
        return p;
    };

    _Float16* h1    = alloc16((long)B*112*112*64);
    _Float16* p1    = alloc16((long)B*37*37*64);
    _Float16* bufA  = alloc16((long)B*36*36*128);
    _Float16* bufB  = alloc16((long)B*36*36*128);
    _Float16* p2    = alloc16((long)B*16*16*128);
    _Float16* bufC  = alloc16((long)B*15*15*256);
    _Float16* bufD  = alloc16((long)B*15*15*256);
    _Float16* feats = alloc16((long)B*12800);
    _Float16* f1o   = alloc16((long)B*4096);
    _Float16* f2o   = alloc16((long)B*4096);
    _Float16* w2h[4];
    w2h[0] = alloc16(128L*4*64);
    for (int i = 1; i < 4; ++i) w2h[i] = alloc16(128L*4*128);
    _Float16* w3h[6];
    w3h[0] = alloc16(256L*4*128);
    for (int i = 1; i < 6; ++i) w3h[i] = alloc16(256L*4*256);
    _Float16* fc1h = alloc16(4096L*12800);
    _Float16* fc2h = alloc16(4096L*4096);
    _Float16* fc3h = alloc16(1000L*4096);

    // stem: conv1 + maxpool 3x3/3
    conv1_k<<<(B*112*112)/256, 256, 0, stream>>>(x, w1, b1, h1);
    {
        long tot = (long)B*37*37*64;
        maxpool_k<<<cdiv_l(tot,256), 256, 0, stream>>>(h1, p1, 112,112, 37,37, 64, 3,3, (int)tot);
    }

    auto conv_layer = [&](const _Float16* in, _Float16* outp,
                          const float* wsrc, const float* bsrc, _Float16* wdst,
                          int Cin, int Cout, int Hi) {
        int wtot = Cout * Cin * 4;
        reorder_w2x2<<<cdiv_l(wtot,256), 256, 0, stream>>>(wsrc, wdst, Cout, Cin, wtot);
        int Ho = Hi - 1, Wo = Hi - 1, Wi = Hi;
        long npix = (long)B * Ho * Wo;           // always a multiple of 32
        int nTiles = (int)(npix >> 5);
        int mTiles = (Cout + 31) >> 5;
        int blocks = mTiles * cdiv_l(nTiles, 8); // 8 N-tiles per block
        if (Cin == 64)
            conv2x2_gemm_t<6><<<blocks, 256, 0, stream>>>(in, wdst, bsrc, outp,
                                                          Hi, Wi, Cout, Ho, Wo, nTiles);
        else if (Cin == 128)
            conv2x2_gemm_t<7><<<blocks, 256, 0, stream>>>(in, wdst, bsrc, outp,
                                                          Hi, Wi, Cout, Ho, Wo, nTiles);
        else
            conv2x2_gemm_t<8><<<blocks, 256, 0, stream>>>(in, wdst, bsrc, outp,
                                                          Hi, Wi, Cout, Ho, Wo, nTiles);
    };

    conv_layer(p1,   bufA, w2[0], b2[0], w2h[0],  64, 128, 37); // ->36
    conv_layer(bufA, bufB, w2[1], b2[1], w2h[1], 128, 128, 36); // ->35
    conv_layer(bufB, bufA, w2[2], b2[2], w2h[2], 128, 128, 35); // ->34
    conv_layer(bufA, bufB, w2[3], b2[3], w2h[3], 128, 128, 34); // ->33
    {
        long tot = (long)B*16*16*128;
        maxpool_k<<<cdiv_l(tot,256), 256, 0, stream>>>(bufB, p2, 33,33, 16,16, 128, 2,2, (int)tot);
    }
    conv_layer(p2,   bufC, w3[0], b3[0], w3h[0], 128, 256, 16); // ->15
    conv_layer(bufC, bufD, w3[1], b3[1], w3h[1], 256, 256, 15); // ->14
    conv_layer(bufD, bufC, w3[2], b3[2], w3h[2], 256, 256, 14); // ->13
    conv_layer(bufC, bufD, w3[3], b3[3], w3h[3], 256, 256, 13); // ->12
    conv_layer(bufD, bufC, w3[4], b3[4], w3h[4], 256, 256, 12); // ->11
    conv_layer(bufC, bufD, w3[5], b3[5], w3h[5], 256, 256, 11); // ->10

    spp_k<<<cdiv_l(64L*50*256,256), 256, 0, stream>>>(bufD, feats);

    auto fc_layer = [&](const _Float16* in, void* outp,
                        const float* wsrc, const float* bsrc, _Float16* wdst,
                        int M, int K, int relu, int of32) {
        long wtot = (long)M * K;
        cvt_f32_f16<<<cdiv_l(wtot,256), 256, 0, stream>>>(wsrc, wdst, wtot);
        int mTiles = (M + 31) >> 5, nTiles = B >> 5;
        int waves  = mTiles * nTiles;
        fc_gemm<<<cdiv_l(waves,8), 256, 0, stream>>>(in, wdst, bsrc, outp, M, K, B, relu, of32);
    };

    fc_layer(feats, f1o,   fc1w, fc1b, fc1h, 4096, 12800, 1, 0);
    fc_layer(f1o,   f2o,   fc2w, fc2b, fc2h, 4096, 4096,  1, 0);
    fc_layer(f2o,   d_out, fc3w, fc3b, fc3h, 1000, 4096,  0, 1);
}
